// E81B_codebook_45990509806224
// MI455X (gfx1250) — compile-verified, matrix operand order checked by reading, not run on hardware
//
#include <hip/hip_runtime.h>

typedef __attribute__((ext_vector_type(16))) _Float16 v16h;
typedef __attribute__((ext_vector_type(8)))  float    v8f;

#define NCODES 256
#define ROWS_PER_WAVE 16
#define WAVES_PER_BLOCK 8
#define ROWS_PER_BLOCK (ROWS_PER_WAVE * WAVES_PER_BLOCK)   // 128

// D = A(16x32 f16) * B(32x16 f16) + 0
// A row m: K=0..7 = x_hi (lane m),  K=8..15 = x_lo (lane m+16),
//          K=16   = 1.0  (lane m),  K=24    = 0.0 (lane m+16), rest 0
// B col n: K=0..7 = g_n,  K=8..15 = g_n (dup),
//          K=16 = K=24 = -||g_n||^2/2 (exact in fp16), rest 0
// => D[m][n] = x[m].g_n - ||g_n||^2/2 == score/2, argmax-equivalent.
__global__ __launch_bounds__(256) void
e8p_nearest_kernel(const float* __restrict__ X,
                   const float* __restrict__ grid,
                   const float* __restrict__ gnorm,
                   float* __restrict__ xq_out,   // [N*8]
                   float* __restrict__ idx_out,  // [N]  (index stored as float)
                   int N)
{
    __shared__ uint4 sLower[NCODES];  // 8 packed fp16 of codeword (exact)
    __shared__ uint4 sUpper[NCODES];  // { fp16(-norm/2), 0,0,0,0,0,0,0 }

    const int tid = threadIdx.x;

    // ---- stage codebook to LDS (one code per thread) ----
    {
        const float4* g4 = (const float4*)(grid + (size_t)tid * 8);
        float4 g0 = g4[0];
        float4 g1 = g4[1];
        union { _Float16 h[8]; uint4 u; } p;
        p.h[0] = (_Float16)g0.x; p.h[1] = (_Float16)g0.y;
        p.h[2] = (_Float16)g0.z; p.h[3] = (_Float16)g0.w;
        p.h[4] = (_Float16)g1.x; p.h[5] = (_Float16)g1.y;
        p.h[6] = (_Float16)g1.z; p.h[7] = (_Float16)g1.w;
        sLower[tid] = p.u;
        union { _Float16 h[8]; uint4 u; } q;
        #pragma unroll
        for (int k = 0; k < 8; ++k) q.h[k] = (_Float16)0.0f;
        q.h[0] = (_Float16)(-0.5f * gnorm[tid]);
        sUpper[tid] = q.u;
    }
    __syncthreads();

    const int wave = tid >> 5;
    const int lane = tid & 31;
    const int rowBase = blockIdx.x * ROWS_PER_BLOCK + wave * ROWS_PER_WAVE;

    // ---- build A fragment ----
    int myRow = rowBase + (lane & 15);
    int ldRow = myRow < N ? myRow : (N - 1);          // clamp, keep EXEC uniform
    const float4* x4 = (const float4*)(X + (size_t)ldRow * 8);
    float4 x0 = x4[0];
    float4 x1 = x4[1];
    float xs[8] = { x0.x, x0.y, x0.z, x0.w, x1.x, x1.y, x1.z, x1.w };

    const bool hiHalf = (lane < 16);
    v16h a;
    #pragma unroll
    for (int k = 0; k < 16; ++k) a[k] = (_Float16)0.0f;
    #pragma unroll
    for (int k = 0; k < 8; ++k) {
        _Float16 h = (_Float16)xs[k];
        _Float16 l = (_Float16)(xs[k] - (float)h);
        a[k] = hiHalf ? h : l;
    }
    a[8] = hiHalf ? (_Float16)1.0f : (_Float16)0.0f;  // K=16 -> 1.0 ; K=24 -> 0

    float best[8];
    int   bidx[8];
    #pragma unroll
    for (int i = 0; i < 8; ++i) { best[i] = -3.0e38f; bidx[i] = 0; }

    const int cLane = lane & 15;

    // ---- 16 tiles; B double-buffered in LDS->VGPR, D double-buffered so the
    //      compare stage runs one tile behind the in-flight WMMA ----
    union Bfrag { uint4 q[2]; v16h v; };
    Bfrag buf[2];
    buf[0].q[0] = sLower[cLane];
    buf[0].q[1] = sUpper[cLane];
    buf[1].q[0] = sLower[cLane + 16];
    buf[1].q[1] = sUpper[cLane + 16];

    v8f d[2];
    d[0] = __builtin_amdgcn_wmma_f32_16x16x32_f16(
               false, a, false, buf[0].v, (short)0, (v8f){}, false, false);

    #pragma unroll
    for (int t = 1; t < 16; ++t) {
        if (t < 15) {                               // prefetch B for tile t+1
            buf[(t + 1) & 1].q[0] = sLower[(t + 1) * 16 + cLane];
            buf[(t + 1) & 1].q[1] = sUpper[(t + 1) * 16 + cLane];
        }

        d[t & 1] = __builtin_amdgcn_wmma_f32_16x16x32_f16(
                       false, a, false, buf[t & 1].v, (short)0, (v8f){}, false, false);

        // compare tile t-1 while WMMA t is in flight
        const int cprev = (t - 1) * 16 + cLane;
        #pragma unroll
        for (int i = 0; i < 8; ++i) {
            float s = d[(t - 1) & 1][i];
            if (s > best[i]) { best[i] = s; bidx[i] = cprev; }
        }
    }
    {   // drain: compare tile 15
        const int clast = 15 * 16 + cLane;
        #pragma unroll
        for (int i = 0; i < 8; ++i) {
            float s = d[1][i];
            if (s > best[i]) { best[i] = s; bidx[i] = clast; }
        }
    }

    // ---- argmax reduction across the 16-lane halves (xor masks stay in-half) ----
    #pragma unroll
    for (int m = 8; m >= 1; m >>= 1) {
        #pragma unroll
        for (int i = 0; i < 8; ++i) {
            float ov = __shfl_xor(best[i], m, 32);
            int   oi = __shfl_xor(bidx[i], m, 32);
            if (ov > best[i] || (ov == best[i] && oi < bidx[i])) {
                best[i] = ov; bidx[i] = oi;
            }
        }
    }

    // ---- writers: lanes 0-7 -> rows 0-7, lanes 16-23 -> rows 8-15 ----
    const int sel = lane & 7;
    int bc = bidx[0];
    #pragma unroll
    for (int i = 1; i < 8; ++i) { if (sel == i) bc = bidx[i]; }

    if ((lane & 15) < 8) {
        const int r = rowBase + sel + ((lane >= 16) ? 8 : 0);
        if (r < N) {
            const float4* gq = (const float4*)(grid + (size_t)bc * 8);
            float4 q0 = gq[0];
            float4 q1 = gq[1];
            float4* o = (float4*)(xq_out + (size_t)r * 8);
            o[0] = q0;
            o[1] = q1;
            idx_out[r] = (float)bc;
        }
    }
}

extern "C" void kernel_launch(void* const* d_in, const int* in_sizes, int n_in,
                              void* d_out, int out_size, void* d_ws, size_t ws_size,
                              hipStream_t stream) {
    const float* X     = (const float*)d_in[0];   // [N,8] f32
    const float* grid  = (const float*)d_in[1];   // [256,8] f32
    const float* gnorm = (const float*)d_in[2];   // [256] f32

    const int N = in_sizes[0] / 8;

    float* xq  = (float*)d_out;                   // first N*8 floats: Xq
    float* idx = (float*)d_out + (size_t)N * 8;   // then N index values

    const int blocks = (N + ROWS_PER_BLOCK - 1) / ROWS_PER_BLOCK;
    e8p_nearest_kernel<<<blocks, 256, 0, stream>>>(X, grid, gnorm, xq, idx, N);
}
